// GroupedQueryAttention_1082331758943
// MI455X (gfx1250) — matrix operand order repaired and checked by reading, not compiled, not run on hardware
//
#include <hip/hip_runtime.h>

// ---------------------------------------------------------------------------
// GQA attention block for MI455X (gfx1250, wave32, WMMA 16x16x32 f16).
// GEMMs: TDM (tensor_load_to_lds) double-buffered LDS staging + WMMA.
// Attention: flash-style causal GQA with WMMA + LDS P-shuffle.
// ---------------------------------------------------------------------------

typedef __attribute__((ext_vector_type(16))) _Float16 v16h;
typedef __attribute__((ext_vector_type(8)))  _Float16 v8h;
typedef __attribute__((ext_vector_type(8)))  float    v8f;
typedef __attribute__((ext_vector_type(4)))  unsigned u32x4;
typedef __attribute__((ext_vector_type(8)))  int      i32x8;
typedef __attribute__((ext_vector_type(4)))  int      i32x4;

union V16 { v16h v; v8h h[2]; };

#define D_MODEL 4096
#define N_HEADS 32
#define N_KV    8
#define HEAD_DIM 128
#define GROUP   4
#define BATCH   2
#define SEQ     2048
#define KV_D    (N_KV * HEAD_DIM)   // 1024

// ---------------------------------------------------------------------------
// TDM: issue a 2-D tile load (f16 elements) from global memory into LDS.
// Descriptor layout per CDNA5 ISA sec. 8.3-8.6 (D# groups 0/1; groups 2/3
// zero => 2-D tile, tile_dim2=0 unused). data_size code 1 => 2 bytes.
// tile_k elements are contiguous in memory (innermost dim), tile_rows rows
// separated by stride_elems; destination is packed row-major in LDS.
// This toolchain (clang-23 / therock-10.0) exposes the 6-arg builtin:
//   (u32x4 g0, i32x8 g1, i32x4 g2, i32x4 g3, i32x8 extra, i32 cpol)
// ---------------------------------------------------------------------------
__device__ __forceinline__ void tdm_load_tile_f16(
    unsigned lds_byte_addr, const void* gaddr,
    unsigned tile_k, unsigned tile_rows,
    unsigned tensor_k, unsigned tensor_rows,
    unsigned long long stride_elems) {
  unsigned long long ga = (unsigned long long)(uintptr_t)gaddr;
  u32x4 g0;
  g0.x = 1u;                                            // count=1 (valid D#)
  g0.y = lds_byte_addr;                                 // LDS dest (bytes)
  g0.z = (unsigned)(ga & 0xFFFFFFFFu);                  // global_addr[31:0]
  g0.w = (unsigned)((ga >> 32) & 0x01FFFFFFu)           // global_addr[56:32]
         | (2u << 30);                                  // type=2 ("image")
  i32x8 g1;
  g1[0] = 0x00010000;                                   // data_size=1 (2B)
  g1[1] = (int)((tensor_k & 0xFFFFu) << 16);            // tensor_dim0[15:0]
  g1[2] = (int)((tensor_k >> 16) |
                ((tensor_rows & 0xFFFFu) << 16));       // dim0 hi | dim1 lo
  g1[3] = (int)((tensor_rows >> 16) |
                ((tile_k & 0xFFFFu) << 16));            // dim1 hi | tile_dim0
  g1[4] = (int)(tile_rows & 0xFFFFu);                   // tile_dim1 (tile_dim2=0)
  g1[5] = (int)(stride_elems & 0xFFFFFFFFull);          // dim0_stride[31:0]
  g1[6] = (int)((stride_elems >> 32) & 0xFFFFull);      // dim0_stride[47:32]
  g1[7] = 0;                                            // dim1_stride unused (2D)
  i32x4 z4;
  z4[0] = 0; z4[1] = 0; z4[2] = 0; z4[3] = 0;
  i32x8 z8;
  z8[0] = 0; z8[1] = 0; z8[2] = 0; z8[3] = 0;
  z8[4] = 0; z8[5] = 0; z8[6] = 0; z8[7] = 0;
  __builtin_amdgcn_tensor_load_to_lds(g0, g1, z4, z4, z8, 0);
}

// ---------------------------------------------------------------------------
// Elementwise f32 -> f16 convert
// ---------------------------------------------------------------------------
__global__ void k_cvt_f16(const float* __restrict__ in, _Float16* __restrict__ out,
                          long long n) {
  long long i = (long long)blockIdx.x * blockDim.x + threadIdx.x;
  if (i < n) out[i] = (_Float16)in[i];
}

// ---------------------------------------------------------------------------
// Weight transpose + convert: W[K,N] f32 -> WT[N,K] f16 (contiguous along K)
// ---------------------------------------------------------------------------
__global__ void k_wtrans(const float* __restrict__ W, _Float16* __restrict__ WT,
                         int K, int N) {
  long long i = (long long)blockIdx.x * blockDim.x + threadIdx.x;
  if (i >= (long long)K * N) return;
  int n = (int)(i % N);
  int k = (int)(i / N);
  WT[(size_t)n * K + k] = (_Float16)W[i];
}

// ---------------------------------------------------------------------------
// TDM-staged WMMA GEMM: C[M,N] f32 = A[M,K] f16 (row-major) * BT[N,K] f16.
// block = 256 threads (8 waves stacked along M); block tile = 128(M) x 64(N).
// Per k-step (32): TDM loads A tile 128x32 and B tile 64x32 into LDS
// (double buffered, 24KB), wave0 waits TENSORcnt, barrier, WMMA from LDS.
// ---------------------------------------------------------------------------
__global__ void k_gemm_wmma(const _Float16* __restrict__ A,
                            const _Float16* __restrict__ BT,
                            float* __restrict__ C,
                            int M, int N, int K) {
  __shared__ _Float16 smA[2][128 * 32];   // 16 KB
  __shared__ _Float16 smB[2][64 * 32];    //  8 KB

  const int lane = threadIdx.x & 31;
  const int wave = threadIdx.x >> 5;      // 0..7
  const int mblk = blockIdx.y * 128;
  const int n0   = blockIdx.x * 64;
  const int mrow = lane & 15;
  const int half = lane >> 4;

  const unsigned ldsA0 = (unsigned)(uintptr_t)(void*)&smA[0][0];
  const unsigned ldsA1 = (unsigned)(uintptr_t)(void*)&smA[1][0];
  const unsigned ldsB0 = (unsigned)(uintptr_t)(void*)&smB[0][0];
  const unsigned ldsB1 = (unsigned)(uintptr_t)(void*)&smB[1][0];

  v8f acc[4];
#pragma unroll
  for (int t = 0; t < 4; ++t)
#pragma unroll
    for (int i = 0; i < 8; ++i) acc[t][i] = 0.0f;

  // prologue: fill buffer 0
  if (wave == 0) {
    tdm_load_tile_f16(ldsA0, A + (size_t)mblk * K, 32, 128,
                      (unsigned)K, (unsigned)M, (unsigned long long)K);
    tdm_load_tile_f16(ldsB0, BT + (size_t)n0 * K, 32, 64,
                      (unsigned)K, (unsigned)N, (unsigned long long)K);
    __builtin_amdgcn_s_wait_tensorcnt(0);
  }
  __syncthreads();

  int buf = 0;
  for (int k0 = 0; k0 < K; k0 += 32) {
    // issue next tile into the other buffer while computing this one
    if (k0 + 32 < K && wave == 0) {
      unsigned la = buf ? ldsA0 : ldsA1;
      unsigned lb = buf ? ldsB0 : ldsB1;
      tdm_load_tile_f16(la, A + (size_t)mblk * K + (k0 + 32), 32, 128,
                        (unsigned)K, (unsigned)M, (unsigned long long)K);
      tdm_load_tile_f16(lb, BT + (size_t)n0 * K + (k0 + 32), 32, 64,
                        (unsigned)K, (unsigned)N, (unsigned long long)K);
    }

    // A fragment for this wave's 16-row slice (ISA A layout: two 16B chunks)
    const _Float16* arow = &smA[buf][(wave * 16 + mrow) * 32];
    V16 a;
    a.h[0] = *(const v8h*)(arow + half * 8);
    a.h[1] = *(const v8h*)(arow + 16 + half * 8);
#pragma unroll
    for (int t = 0; t < 4; ++t) {
      const _Float16* bcol = &smB[buf][(t * 16 + mrow) * 32];
      V16 b;
      b.h[0] = *(const v8h*)(bcol + half * 8);
      b.h[1] = *(const v8h*)(bcol + 16 + half * 8);
      acc[t] = __builtin_amdgcn_wmma_f32_16x16x32_f16(
          false, a.v, false, b.v, (short)0, acc[t], false, false);
    }

    if (wave == 0) __builtin_amdgcn_s_wait_tensorcnt(0);
    __syncthreads();
    buf ^= 1;
  }

#pragma unroll
  for (int t = 0; t < 4; ++t)
#pragma unroll
    for (int j = 0; j < 8; ++j) {
      int row = mblk + wave * 16 + j + 8 * half;   // C layout: m = j + 8*half
      int col = n0 + t * 16 + mrow;                // n = lane&15
      C[(size_t)row * N + col] = acc[t][j];
    }
}

// ---------------------------------------------------------------------------
// RoPE for Q: f_Q[b,s,h*128+d] f32 -> h_Q same layout f16
// ---------------------------------------------------------------------------
__global__ void k_rope_q(const float* __restrict__ Qf, _Float16* __restrict__ Qh) {
  long long i = (long long)blockIdx.x * blockDim.x + threadIdx.x;
  const long long total = (long long)BATCH * SEQ * N_HEADS * 64;
  if (i >= total) return;
  int d = (int)(i & 63);
  long long rest = i >> 6;
  int h = (int)(rest % N_HEADS);
  long long bs = rest / N_HEADS;           // b*SEQ + s
  int s = (int)(bs % SEQ);
  size_t base = (size_t)bs * D_MODEL + (size_t)h * HEAD_DIM;
  float inv = expf(-(float)(2 * d) * 9.210340371976184f / 128.0f); // 10000^(-2d/128)
  float ang = (float)s * inv;
  float c = cosf(ang), sn = sinf(ang);
  float x1 = Qf[base + d];
  float x2 = Qf[base + d + 64];
  Qh[base + d]      = (_Float16)(x1 * c - x2 * sn);
  Qh[base + d + 64] = (_Float16)(x2 * c + x1 * sn);
}

// ---------------------------------------------------------------------------
// RoPE for K: f_K[b,s,kv*128+d] f32 -> h_K[b,kv,s,d] f16
// ---------------------------------------------------------------------------
__global__ void k_rope_k(const float* __restrict__ Kf, _Float16* __restrict__ Kh) {
  long long i = (long long)blockIdx.x * blockDim.x + threadIdx.x;
  const long long total = (long long)BATCH * SEQ * N_KV * 64;
  if (i >= total) return;
  int d = (int)(i & 63);
  long long rest = i >> 6;
  int kv = (int)(rest % N_KV);
  long long bs = rest / N_KV;
  int s = (int)(bs % SEQ);
  int b = (int)(bs / SEQ);
  size_t ib = (size_t)bs * KV_D + (size_t)kv * HEAD_DIM;
  size_t ob = (((size_t)b * N_KV + kv) * SEQ + s) * HEAD_DIM;
  float inv = expf(-(float)(2 * d) * 9.210340371976184f / 128.0f);
  float ang = (float)s * inv;
  float c = cosf(ang), sn = sinf(ang);
  float x1 = Kf[ib + d];
  float x2 = Kf[ib + d + 64];
  Kh[ob + d]      = (_Float16)(x1 * c - x2 * sn);
  Kh[ob + d + 64] = (_Float16)(x2 * c + x1 * sn);
}

// ---------------------------------------------------------------------------
// V convert+transpose: f_V[b,s,kv*128+dh] f32 -> h_V[b,kv,dh,s] f16
// ---------------------------------------------------------------------------
__global__ void k_vtrans(const float* __restrict__ Vf, _Float16* __restrict__ Vh) {
  long long i = (long long)blockIdx.x * blockDim.x + threadIdx.x;
  const long long total = (long long)BATCH * SEQ * KV_D;
  if (i >= total) return;
  int c = (int)(i % KV_D);
  long long bs = i / KV_D;
  int kv = c >> 7;
  int dh = c & 127;
  int s = (int)(bs % SEQ);
  int b = (int)(bs / SEQ);
  Vh[(((size_t)b * N_KV + kv) * HEAD_DIM + dh) * SEQ + s] = (_Float16)Vf[i];
}

// ---------------------------------------------------------------------------
// Flash-style causal GQA attention.
//   Qh: [B,S,H*128] f16   Kh: [B,KV,S,128] f16   Vh: [B,KV,128,S] f16
//   O : [B,S,H*128] f32
// block = 128 threads (4 waves); each wave owns one 16-query tile.
// ---------------------------------------------------------------------------
__global__ void k_gqa_flash(const _Float16* __restrict__ Qh,
                            const _Float16* __restrict__ Kh,
                            const _Float16* __restrict__ Vh,
                            float* __restrict__ O) {
  __shared__ _Float16 ldsP[4][16 * 32];

  const int lane = threadIdx.x & 31;
  const int wave = threadIdx.x >> 5;
  const int qt = blockIdx.x * 4 + wave;
  const int h  = blockIdx.y;
  const int b  = blockIdx.z;
  const int kv = h / GROUP;
  const int q0 = qt * 16;
  const int m = lane & 15;
  const int half = lane >> 4;
  const float scale = 0.08838834764831845f;   // 1/sqrt(128)
  const float NEG_INF = -__builtin_inff();

  V16 aq[4];
  const _Float16* qrow = Qh + (size_t)(b * SEQ + q0 + m) * D_MODEL + (size_t)h * HEAD_DIM;
#pragma unroll
  for (int c = 0; c < 4; ++c) {
    aq[c].h[0] = *(const v8h*)(qrow + c * 32 + half * 8);
    aq[c].h[1] = *(const v8h*)(qrow + c * 32 + 16 + half * 8);
  }

  v8f o[8];
#pragma unroll
  for (int nt = 0; nt < 8; ++nt)
#pragma unroll
    for (int j = 0; j < 8; ++j) o[nt][j] = 0.0f;
  float mrun[8], lrun[8];
#pragma unroll
  for (int j = 0; j < 8; ++j) { mrun[j] = NEG_INF; lrun[j] = 0.0f; }

  const _Float16* kbase = Kh + ((size_t)b * N_KV + kv) * SEQ * HEAD_DIM;
  const _Float16* vbase = Vh + ((size_t)b * N_KV + kv) * HEAD_DIM * SEQ;
  _Float16* lp = ldsP[wave];

  const int kend = q0 + 16;
  for (int c0 = 0; c0 < kend; c0 += 32) {
    v8f s[2];
#pragma unroll
    for (int t = 0; t < 2; ++t) {
#pragma unroll
      for (int i = 0; i < 8; ++i) s[t][i] = 0.0f;
      const _Float16* krow = kbase + (size_t)(c0 + t * 16 + m) * HEAD_DIM;
#pragma unroll
      for (int c = 0; c < 4; ++c) {
        V16 bk;
        bk.h[0] = *(const v8h*)(krow + c * 32 + half * 8);
        bk.h[1] = *(const v8h*)(krow + c * 32 + 16 + half * 8);
        s[t] = __builtin_amdgcn_wmma_f32_16x16x32_f16(
            false, aq[c].v, false, bk.v, (short)0, s[t], false, false);
      }
    }

    float p0[8], p1[8];
#pragma unroll
    for (int j = 0; j < 8; ++j) {
      int qg = q0 + j + 8 * half;
      float x0 = (c0 + m      <= qg) ? s[0][j] * scale : NEG_INF;
      float x1 = (c0 + 16 + m <= qg) ? s[1][j] * scale : NEG_INF;
      float vmax = fmaxf(x0, x1);
      vmax = fmaxf(vmax, __shfl_xor(vmax, 1, 16));
      vmax = fmaxf(vmax, __shfl_xor(vmax, 2, 16));
      vmax = fmaxf(vmax, __shfl_xor(vmax, 4, 16));
      vmax = fmaxf(vmax, __shfl_xor(vmax, 8, 16));
      float nm = fmaxf(mrun[j], vmax);
      float e0 = __expf(x0 - nm);
      float e1 = __expf(x1 - nm);
      float rs = e0 + e1;
      rs += __shfl_xor(rs, 1, 16);
      rs += __shfl_xor(rs, 2, 16);
      rs += __shfl_xor(rs, 4, 16);
      rs += __shfl_xor(rs, 8, 16);
      float corr = __expf(mrun[j] - nm);
      lrun[j] = lrun[j] * corr + rs;
      mrun[j] = nm;
#pragma unroll
      for (int nt = 0; nt < 8; ++nt) o[nt][j] *= corr;
      p0[j] = e0;
      p1[j] = e1;
    }

#pragma unroll
    for (int j = 0; j < 8; ++j) {
      int r = j + 8 * half;
      lp[r * 32 + m]      = (_Float16)p0[j];
      lp[r * 32 + 16 + m] = (_Float16)p1[j];
    }
    asm volatile("s_wait_dscnt 0" ::: "memory");
    V16 ap;
    ap.h[0] = *(const v8h*)(lp + m * 32 + half * 8);
    ap.h[1] = *(const v8h*)(lp + m * 32 + 16 + half * 8);

#pragma unroll
    for (int nt = 0; nt < 8; ++nt) {
      const _Float16* vcol = vbase + (size_t)(nt * 16 + m) * SEQ + c0;
      V16 bv;
      bv.h[0] = *(const v8h*)(vcol + half * 8);
      bv.h[1] = *(const v8h*)(vcol + 16 + half * 8);
      o[nt] = __builtin_amdgcn_wmma_f32_16x16x32_f16(
          false, ap.v, false, bv.v, (short)0, o[nt], false, false);
    }
  }

  float* orow = O + (size_t)(b * SEQ + q0) * D_MODEL + (size_t)h * HEAD_DIM;
#pragma unroll
  for (int j = 0; j < 8; ++j) {
    float invl = 1.0f / lrun[j];
    int r = j + 8 * half;
#pragma unroll
    for (int nt = 0; nt < 8; ++nt)
      orow[(size_t)r * D_MODEL + nt * 16 + m] = o[nt][j] * invl;
  }
}

// ---------------------------------------------------------------------------
// Host launcher
// ---------------------------------------------------------------------------
extern "C" void kernel_launch(void* const* d_in, const int* in_sizes, int n_in,
                              void* d_out, int out_size, void* d_ws, size_t ws_size,
                              hipStream_t stream) {
  (void)in_sizes; (void)n_in; (void)out_size; (void)ws_size;
  const float* x  = (const float*)d_in[0];
  // d_in[1] is the causal mask; the kernel applies causality analytically.
  const float* Wq = (const float*)d_in[2];
  const float* Wk = (const float*)d_in[3];
  const float* Wv = (const float*)d_in[4];
  const float* Wo = (const float*)d_in[5];
  float* out = (float*)d_out;

  const size_t BS = (size_t)BATCH * SEQ;           // 4096
  const size_t XD = BS * D_MODEL;                  // 16.8M
  const size_t KD = BS * KV_D;                     // 4.2M

  char* ws = (char*)d_ws;
  size_t off = 0;
  auto bump = [&](size_t bytes) -> void* {
    void* p = ws + off;
    off = (off + bytes + 255) & ~(size_t)255;
    return p;
  };
  _Float16* h_x   = (_Float16*)bump(XD * 2);
  _Float16* h_WqT = (_Float16*)bump((size_t)D_MODEL * D_MODEL * 2);
  _Float16* h_WkT = (_Float16*)bump((size_t)D_MODEL * KV_D * 2);
  _Float16* h_WvT = (_Float16*)bump((size_t)D_MODEL * KV_D * 2);
  _Float16* h_WoT = (_Float16*)bump((size_t)D_MODEL * D_MODEL * 2);
  float*    f_Q   = (float*)bump(XD * 4);
  float*    f_K   = (float*)bump(KD * 4);
  float*    f_V   = (float*)bump(KD * 4);
  _Float16* h_Q   = (_Float16*)bump(XD * 2);
  _Float16* h_K   = (_Float16*)bump(KD * 2);
  _Float16* h_V   = (_Float16*)bump(KD * 2);
  float*    f_att = (float*)bump(XD * 4);
  _Float16* h_att = (_Float16*)bump(XD * 2);

  const int TPB = 256;
  auto blocks = [&](long long n) { return (unsigned)((n + TPB - 1) / TPB); };

  // 1. convert inputs / weights (weights transposed to [N,K] f16)
  k_cvt_f16<<<blocks((long long)XD), TPB, 0, stream>>>(x, h_x, (long long)XD);
  k_wtrans<<<blocks((long long)D_MODEL * D_MODEL), TPB, 0, stream>>>(Wq, h_WqT, D_MODEL, D_MODEL);
  k_wtrans<<<blocks((long long)D_MODEL * KV_D),   TPB, 0, stream>>>(Wk, h_WkT, D_MODEL, KV_D);
  k_wtrans<<<blocks((long long)D_MODEL * KV_D),   TPB, 0, stream>>>(Wv, h_WvT, D_MODEL, KV_D);
  k_wtrans<<<blocks((long long)D_MODEL * D_MODEL), TPB, 0, stream>>>(Wo, h_WoT, D_MODEL, D_MODEL);

  // 2. QKV projections (TDM-staged WMMA GEMMs); block tile 128M x 64N
  {
    dim3 g(D_MODEL / 64, (unsigned)(BS / 128));
    k_gemm_wmma<<<g, 256, 0, stream>>>(h_x, h_WqT, f_Q, (int)BS, D_MODEL, D_MODEL);
  }
  {
    dim3 g(KV_D / 64, (unsigned)(BS / 128));
    k_gemm_wmma<<<g, 256, 0, stream>>>(h_x, h_WkT, f_K, (int)BS, KV_D, D_MODEL);
    k_gemm_wmma<<<g, 256, 0, stream>>>(h_x, h_WvT, f_V, (int)BS, KV_D, D_MODEL);
  }

  // 3. RoPE (+f16 convert, +layout change for K) and V transpose
  k_rope_q<<<blocks((long long)BATCH * SEQ * N_HEADS * 64), TPB, 0, stream>>>(f_Q, h_Q);
  k_rope_k<<<blocks((long long)BATCH * SEQ * N_KV * 64),    TPB, 0, stream>>>(f_K, h_K);
  k_vtrans<<<blocks((long long)KD), TPB, 0, stream>>>(f_V, h_V);

  // 4. causal GQA flash attention
  {
    dim3 g(SEQ / 64, N_HEADS, BATCH);
    k_gqa_flash<<<g, 128, 0, stream>>>(h_Q, h_K, h_V, f_att);
  }

  // 5. output projection
  k_cvt_f16<<<blocks((long long)XD), TPB, 0, stream>>>(f_att, h_att, (long long)XD);
  {
    dim3 g(D_MODEL / 64, (unsigned)(BS / 128));
    k_gemm_wmma<<<g, 256, 0, stream>>>(h_att, h_WoT, out, (int)BS, D_MODEL, D_MODEL);
  }
}
